// Attention_C_56736517980279
// MI455X (gfx1250) — matrix operand order repaired and checked by reading, compiled, not run on hardware
//
#include <hip/hip_runtime.h>
#include <hip/hip_bf16.h>
#include <math.h>

typedef __attribute__((ext_vector_type(16))) _Float16 v16h;
typedef __attribute__((ext_vector_type(8)))  _Float16 v8h;
typedef __attribute__((ext_vector_type(8)))  float    v8f;

#define WMMA_F16(A, B, C) \
  __builtin_amdgcn_wmma_f32_16x16x32_f16(false, (A), false, (B), (short)0, (C), false, false)

// CDNA5 async memory->LDS copy (ASYNCcnt-tracked), per CDNA5 ISA
// s15.18.3 / CDNA5 ISA s4. %0 = LDS byte offset VGPR, %1 = 64-bit
// global address VGPR pair.
#define ASYNC_LOAD_B128(loff, gaddr)                                      \
  asm volatile("global_load_async_to_lds_b128 %0, %1, off"                \
               :: "v"(loff), "v"(gaddr) : "memory")
#define WAIT_ASYNC0() asm volatile("s_wait_asynccnt 0x0" ::: "memory")

// ---------------------------------------------------------------------------
// Elementwise f32 -> f16 conversion (weights)
// ---------------------------------------------------------------------------
__global__ __launch_bounds__(256) void k_cvt_f16(const float* __restrict__ in,
                                                 _Float16* __restrict__ out, int n) {
  int i = blockIdx.x * 256 + threadIdx.x;
  if (i < n) out[i] = (_Float16)in[i];
}

// ---------------------------------------------------------------------------
// Tiled transpose + f16 cast: x[b][c][n] f32 -> xt[b][n][c] f16 (64x64 tiles)
// ---------------------------------------------------------------------------
__global__ __launch_bounds__(256) void k_transpose(const float* __restrict__ x,
                                                   _Float16* __restrict__ xt) {
  __shared__ _Float16 tile[64][72];
  int b = blockIdx.z, c0 = blockIdx.y * 64, n0 = blockIdx.x * 64;
  int t = threadIdx.x, tn = t & 63, tc4 = t >> 6;  // 0..3
#pragma unroll
  for (int p = 0; p < 16; ++p) {
    int c = p * 4 + tc4;
    tile[tn][c] = (_Float16)x[((size_t)b * 384 + c0 + c) * 4096 + n0 + tn];
  }
  __syncthreads();
#pragma unroll
  for (int p = 0; p < 16; ++p) {
    int n = p * 4 + tc4;
    xt[((size_t)b * 4096 + n0 + n) * 384 + c0 + tn] = tile[n][tn];
  }
}

// ---------------------------------------------------------------------------
// QKV 1x1-conv GEMM with async-to-LDS double-buffered pipeline.
// C[b,m,n] = sum_k A[m,k] * Xt[b,n,k] + bias[m]   (M=1152, N=4096, K=384)
// Block tile 128x128, 8 waves (4m x 2n), wave tile 32x64, K-step 32.
// Both A and B tiles are row-major-in-K => staged purely with
// global_load_async_to_lds_b128; prefetch of tile k+1 overlaps WMMA of tile k.
// Dynamic LDS layout (f16 units): Abuf[p] @ p*5120, Bbuf[p] @ 10240+p*5120
// (row stride 40 f16 = 80B: 16B-aligned, bank-conflict-free b128 frag loads).
// ---------------------------------------------------------------------------
__global__ __launch_bounds__(256) void k_gemm_qkv(const _Float16* __restrict__ A,
                                                  const _Float16* __restrict__ Xt,
                                                  const float* __restrict__ bias,
                                                  _Float16* __restrict__ C) {
  constexpr int ST = 40;
  extern __shared__ _Float16 smem[];

  const int b  = blockIdx.z;
  const int mB = blockIdx.y * 128;
  const int nB = blockIdx.x * 128;
  const _Float16* Bg = Xt + (size_t)b * 4096 * 384;

  const int t    = threadIdx.x;
  const int lane = t & 31;
  const int wave = t >> 5;
  const int wm   = wave >> 1;   // 0..3
  const int wn   = wave & 1;    // 0..1
  const int half = lane >> 4;
  const int l16  = lane & 15;

  v8f acc[2][4];
#pragma unroll
  for (int i = 0; i < 2; ++i)
#pragma unroll
    for (int j = 0; j < 4; ++j)
#pragma unroll
      for (int e = 0; e < 8; ++e) acc[i][j][e] = 0.0f;

  // Issue this wave's share of async copies for one 128x32 A tile and one
  // 128x32 B tile into buffer p (4 async b128 per wave; 512B per instr).
  auto issue = [&](int k0, int p) {
#pragma unroll
    for (int q = 0; q < 2; ++q) {
      int chunk = wave * 64 + q * 32 + lane;         // 0..511, 8 f16 each
      int row = chunk >> 2, off = (chunk & 3) * 8;
      unsigned int la = (unsigned int)((p * 5120 + row * ST + off) * 2);
      unsigned long long ga =
          (unsigned long long)(uintptr_t)(A + (size_t)(mB + row) * 384 + k0 + off);
      ASYNC_LOAD_B128(la, ga);
      unsigned int lb = (unsigned int)((10240 + p * 5120 + row * ST + off) * 2);
      unsigned long long gb =
          (unsigned long long)(uintptr_t)(Bg + (size_t)(nB + row) * 384 + k0 + off);
      ASYNC_LOAD_B128(lb, gb);
    }
  };

  issue(0, 0);
  for (int kt = 0; kt < 12; ++kt) {
    const int p = kt & 1;
    WAIT_ASYNC0();        // this wave's copies for tile kt are in LDS
    __syncthreads();      // everyone's copies visible; prev reads of p^1 done
    if (kt < 11) issue((kt + 1) * 32, p ^ 1);  // overlaps with WMMA below

    const _Float16* As = smem + p * 5120;
    const _Float16* Bt = smem + 10240 + p * 5120;
    v16h af[2], bf[4];
#pragma unroll
    for (int i = 0; i < 2; ++i) {
      int m = wm * 32 + i * 16 + l16;
      v8h lo = *(const v8h*)(&As[m * ST + half * 8]);
      v8h hi = *(const v8h*)(&As[m * ST + 16 + half * 8]);
#pragma unroll
      for (int e = 0; e < 8; ++e) { af[i][e] = lo[e]; af[i][e + 8] = hi[e]; }
    }
#pragma unroll
    for (int j = 0; j < 4; ++j) {
      int n = wn * 64 + j * 16 + l16;
      v8h lo = *(const v8h*)(&Bt[n * ST + half * 8]);
      v8h hi = *(const v8h*)(&Bt[n * ST + 16 + half * 8]);
#pragma unroll
      for (int e = 0; e < 8; ++e) { bf[j][e] = lo[e]; bf[j][e + 8] = hi[e]; }
    }
#pragma unroll
    for (int i = 0; i < 2; ++i)
#pragma unroll
      for (int j = 0; j < 4; ++j) acc[i][j] = WMMA_F16(af[i], bf[j], acc[i][j]);
  }

#pragma unroll
  for (int i = 0; i < 2; ++i)
#pragma unroll
    for (int j = 0; j < 4; ++j)
#pragma unroll
      for (int r = 0; r < 8; ++r) {
        int m = mB + wm * 32 + i * 16 + half * 8 + r;
        int n = nB + wn * 64 + j * 16 + l16;
        C[((size_t)b * 1152 + m) * 4096 + n] = (_Float16)(acc[i][j][r] + bias[m]);
      }
}

// ---------------------------------------------------------------------------
// Generic conv-as-GEMM (B transpose-staged): C[b,m,n] = A[m,:].B[b,:,n]+bias
// Used for the final 1x1 conv (f32 output). Block 128x128, wave 32x64.
// ---------------------------------------------------------------------------
template <bool OUT_F32>
__global__ __launch_bounds__(256) void k_gemm_ck(const _Float16* __restrict__ A,
                                                 const _Float16* __restrict__ Bb,
                                                 const float* __restrict__ bias,
                                                 void* __restrict__ Cout,
                                                 int M, int N, int K) {
  constexpr int ST = 40;
  __shared__ _Float16 As[128 * ST];
  __shared__ _Float16 Bt[128 * ST];

  const int b  = blockIdx.z;
  const int mB = blockIdx.y * 128;
  const int nB = blockIdx.x * 128;
  const _Float16* B = Bb + (size_t)b * K * N;

  const int t = threadIdx.x, lane = t & 31, wave = t >> 5;
  const int wm = wave >> 1, wn = wave & 1;
  const int half = lane >> 4, l16 = lane & 15;

  v8f acc[2][4];
#pragma unroll
  for (int i = 0; i < 2; ++i)
#pragma unroll
    for (int j = 0; j < 4; ++j)
#pragma unroll
      for (int e = 0; e < 8; ++e) acc[i][j][e] = 0.0f;

  for (int k0 = 0; k0 < K; k0 += 32) {
#pragma unroll
    for (int p = 0; p < 2; ++p) {
      int chunk = t + p * 256;
      int row = chunk >> 2, off = (chunk & 3) * 8;
      v8h va = *(const v8h*)(A + (size_t)(mB + row) * K + k0 + off);
      *(v8h*)(&As[row * ST + off]) = va;
    }
#pragma unroll
    for (int p = 0; p < 2; ++p) {
      int chunk = t + p * 256;
      int kr = chunk >> 4, nc = (chunk & 15) * 8;
      v8h vb = *(const v8h*)(B + (size_t)(k0 + kr) * N + nB + nc);
#pragma unroll
      for (int j = 0; j < 8; ++j) Bt[(nc + j) * ST + kr] = vb[j];
    }
    __syncthreads();

    v16h af[2], bf[4];
#pragma unroll
    for (int i = 0; i < 2; ++i) {
      int m = wm * 32 + i * 16 + l16;
      v8h lo = *(const v8h*)(&As[m * ST + half * 8]);
      v8h hi = *(const v8h*)(&As[m * ST + 16 + half * 8]);
#pragma unroll
      for (int e = 0; e < 8; ++e) { af[i][e] = lo[e]; af[i][e + 8] = hi[e]; }
    }
#pragma unroll
    for (int j = 0; j < 4; ++j) {
      int n = wn * 64 + j * 16 + l16;
      v8h lo = *(const v8h*)(&Bt[n * ST + half * 8]);
      v8h hi = *(const v8h*)(&Bt[n * ST + 16 + half * 8]);
#pragma unroll
      for (int e = 0; e < 8; ++e) { bf[j][e] = lo[e]; bf[j][e + 8] = hi[e]; }
    }
#pragma unroll
    for (int i = 0; i < 2; ++i)
#pragma unroll
      for (int j = 0; j < 4; ++j) acc[i][j] = WMMA_F16(af[i], bf[j], acc[i][j]);
    __syncthreads();
  }

#pragma unroll
  for (int i = 0; i < 2; ++i)
#pragma unroll
    for (int j = 0; j < 4; ++j)
#pragma unroll
      for (int r = 0; r < 8; ++r) {
        int m = mB + wm * 32 + i * 16 + half * 8 + r;
        int n = nB + wn * 64 + j * 16 + l16;
        float v = acc[i][j][r] + bias[m];
        if (OUT_F32)
          ((float*)Cout)[((size_t)b * M + m) * N + n] = v;
        else
          ((_Float16*)Cout)[((size_t)b * M + m) * N + n] = (_Float16)v;
      }
}

// ---------------------------------------------------------------------------
// Depthwise 3x3 conv (pad=1) + bias, f16 in/out, f32 accumulate
// ---------------------------------------------------------------------------
__global__ __launch_bounds__(256) void k_dwconv(const _Float16* __restrict__ in,
                                                const float* __restrict__ Wdw,
                                                const float* __restrict__ bdw,
                                                _Float16* __restrict__ out) {
  size_t i = (size_t)blockIdx.x * 256 + threadIdx.x;  // 16*1152*4096 total
  int n  = (int)(i & 4095);
  int ch = (int)((i >> 12) % 1152);
  int y = n >> 6, x = n & 63;
  const _Float16* base = in + (i - n);
  const float* w = Wdw + (size_t)ch * 9;
  float acc = bdw[ch];
#pragma unroll
  for (int dy = -1; dy <= 1; ++dy)
#pragma unroll
    for (int dx = -1; dx <= 1; ++dx) {
      int yy = y + dy, xx = x + dx;
      if (yy >= 0 && yy < 64 && xx >= 0 && xx < 64)
        acc += (float)base[yy * 64 + xx] * w[(dy + 1) * 3 + (dx + 1)];
    }
  out[i] = (_Float16)acc;
}

// ---------------------------------------------------------------------------
// In-place L2 row normalization over n=4096 (q and k channels only)
// ---------------------------------------------------------------------------
__global__ __launch_bounds__(256) void k_l2norm(_Float16* __restrict__ dwh) {
  __shared__ float red[256];
  int row = blockIdx.x;
  int b = row / 768, ch = row % 768;
  _Float16* p = dwh + ((size_t)b * 1152 + ch) * 4096;
  int t = threadIdx.x;
  float ss = 0.0f;
  for (int i = t; i < 4096; i += 256) { float v = (float)p[i]; ss += v * v; }
  red[t] = ss; __syncthreads();
  for (int s = 128; s > 0; s >>= 1) { if (t < s) red[t] += red[t + s]; __syncthreads(); }
  float inv = 1.0f / fmaxf(sqrtf(red[0]), 1e-12f);
  for (int i = t; i < 4096; i += 256) p[i] = (_Float16)((float)p[i] * inv);
}

// ---------------------------------------------------------------------------
// attn[bh,c,d] = temp[h] * sum_n q[c,n]*k[d,n]   (M=48->64, N=48, K=4096)
// ---------------------------------------------------------------------------
__global__ __launch_bounds__(128) void k_attn_qk(const _Float16* __restrict__ dwh,
                                                 const float* __restrict__ temp,
                                                 float* __restrict__ attn) {
  constexpr int ST = 72;
  __shared__ _Float16 Aq[64 * ST];
  __shared__ _Float16 Bk[48 * ST];

  const int bh = blockIdx.x;
  const int b = bh >> 3, h = bh & 7;
  const _Float16* Q = dwh + ((size_t)b * 1152 + h * 48) * 4096;
  const _Float16* Kp = Q + (size_t)384 * 4096;

  const int t = threadIdx.x, lane = t & 31, wave = t >> 5;
  const int half = lane >> 4, l16 = lane & 15;

  for (int i = t; i < 16 * ST; i += 128) Aq[48 * ST + i] = (_Float16)0;

  v8f acc[3];
#pragma unroll
  for (int j = 0; j < 3; ++j)
#pragma unroll
    for (int e = 0; e < 8; ++e) acc[j][e] = 0.0f;

  for (int k0 = 0; k0 < 4096; k0 += 64) {
#pragma unroll
    for (int p = 0; p < 3; ++p) {
      int chunk = t + p * 128;
      int row = chunk >> 3, off = (chunk & 7) * 8;
      *(v8h*)(&Aq[row * ST + off]) = *(const v8h*)(Q + (size_t)row * 4096 + k0 + off);
      *(v8h*)(&Bk[row * ST + off]) = *(const v8h*)(Kp + (size_t)row * 4096 + k0 + off);
    }
    __syncthreads();
#pragma unroll
    for (int ks = 0; ks < 2; ++ks) {
      v16h af;
      {
        int m = wave * 16 + l16;
        v8h lo = *(const v8h*)(&Aq[m * ST + ks * 32 + half * 8]);
        v8h hi = *(const v8h*)(&Aq[m * ST + ks * 32 + 16 + half * 8]);
#pragma unroll
        for (int e = 0; e < 8; ++e) { af[e] = lo[e]; af[e + 8] = hi[e]; }
      }
#pragma unroll
      for (int j = 0; j < 3; ++j) {
        int d = j * 16 + l16;
        v16h bf;
        v8h lo = *(const v8h*)(&Bk[d * ST + ks * 32 + half * 8]);
        v8h hi = *(const v8h*)(&Bk[d * ST + ks * 32 + 16 + half * 8]);
#pragma unroll
        for (int e = 0; e < 8; ++e) { bf[e] = lo[e]; bf[e + 8] = hi[e]; }
        acc[j] = WMMA_F16(af, bf, acc[j]);
      }
    }
    __syncthreads();
  }

  float th = temp[h];
#pragma unroll
  for (int j = 0; j < 3; ++j)
#pragma unroll
    for (int r = 0; r < 8; ++r) {
      int c = wave * 16 + half * 8 + r;
      int d = j * 16 + l16;
      if (c < 48) attn[((size_t)bh * 48 + c) * 48 + d] = acc[j][r] * th;
    }
}

// ---------------------------------------------------------------------------
// Softmax over d (48) -> f16, written into zero-padded 64x64 tiles per (b,h)
// ---------------------------------------------------------------------------
__global__ __launch_bounds__(64) void k_softmax48(const float* __restrict__ attn,
                                                  _Float16* __restrict__ attnh) {
  __shared__ float red[64];
  int idx = blockIdx.x;
  int bh = idx >> 6, c = idx & 63;
  int t = threadIdx.x;
  _Float16* out = attnh + ((size_t)bh * 64 + c) * 64;
  if (c >= 48) { out[t] = (_Float16)0; return; }
  const float* row = attn + ((size_t)bh * 48 + c) * 48;
  float v = (t < 48) ? row[t] : -1e30f;
  red[t] = v; __syncthreads();
  for (int s = 32; s > 0; s >>= 1) { if (t < s) red[t] = fmaxf(red[t], red[t + s]); __syncthreads(); }
  float mx = red[0]; __syncthreads();
  float e = (t < 48) ? __expf(v - mx) : 0.0f;
  red[t] = e; __syncthreads();
  for (int s = 32; s > 0; s >>= 1) { if (t < s) red[t] += red[t + s]; __syncthreads(); }
  out[t] = (t < 48) ? (_Float16)(e / red[0]) : (_Float16)0;
}

// ---------------------------------------------------------------------------
// out[c,n] = sum_d attn[c,d] * v[d,n]   (M=48->64, K=48->64, N=4096)
// ---------------------------------------------------------------------------
__global__ __launch_bounds__(256) void k_attn_v(const _Float16* __restrict__ attnh,
                                                const _Float16* __restrict__ dwh,
                                                _Float16* __restrict__ outh) {
  constexpr int ST = 72;
  __shared__ _Float16 As[64 * ST];
  __shared__ _Float16 Bt[128 * ST];

  const int bh = blockIdx.y;
  const int b = bh >> 3, h = bh & 7;
  const _Float16* A = attnh + (size_t)bh * 64 * 64;
  const _Float16* V = dwh + ((size_t)b * 1152 + 768 + h * 48) * 4096;
  const int nB = blockIdx.x * 128;

  const int t = threadIdx.x, lane = t & 31, wave = t >> 5;
  const int wm = wave >> 2, wn = wave & 3;
  const int half = lane >> 4, l16 = lane & 15;

#pragma unroll
  for (int p = 0; p < 2; ++p) {
    int chunk = t + p * 256;
    int row = chunk >> 3, off = (chunk & 7) * 8;
    *(v8h*)(&As[row * ST + off]) = *(const v8h*)(A + (size_t)row * 64 + off);
  }
#pragma unroll
  for (int p = 0; p < 3; ++p) {
    int chunk = t + p * 256;
    int d = chunk >> 4, nc = (chunk & 15) * 8;
    v8h vv = *(const v8h*)(V + (size_t)d * 4096 + nB + nc);
#pragma unroll
    for (int j = 0; j < 8; ++j) Bt[(nc + j) * ST + d] = vv[j];
  }
  {
    v8h z;
#pragma unroll
    for (int e = 0; e < 8; ++e) z[e] = (_Float16)0;
    int nn = t >> 1, koff = 48 + (t & 1) * 8;
    *(v8h*)(&Bt[nn * ST + koff]) = z;
  }
  __syncthreads();

  v8f acc[2][2];
#pragma unroll
  for (int i = 0; i < 2; ++i)
#pragma unroll
    for (int j = 0; j < 2; ++j)
#pragma unroll
      for (int e = 0; e < 8; ++e) acc[i][j][e] = 0.0f;

#pragma unroll
  for (int ks = 0; ks < 2; ++ks) {
    v16h af[2], bf[2];
#pragma unroll
    for (int i = 0; i < 2; ++i) {
      int m = wm * 32 + i * 16 + l16;
      v8h lo = *(const v8h*)(&As[m * ST + ks * 32 + half * 8]);
      v8h hi = *(const v8h*)(&As[m * ST + ks * 32 + 16 + half * 8]);
#pragma unroll
      for (int e = 0; e < 8; ++e) { af[i][e] = lo[e]; af[i][e + 8] = hi[e]; }
    }
#pragma unroll
    for (int j = 0; j < 2; ++j) {
      int n = wn * 32 + j * 16 + l16;
      v8h lo = *(const v8h*)(&Bt[n * ST + ks * 32 + half * 8]);
      v8h hi = *(const v8h*)(&Bt[n * ST + ks * 32 + 16 + half * 8]);
#pragma unroll
      for (int e = 0; e < 8; ++e) { bf[j][e] = lo[e]; bf[j][e + 8] = hi[e]; }
    }
#pragma unroll
    for (int i = 0; i < 2; ++i)
#pragma unroll
      for (int j = 0; j < 2; ++j) acc[i][j] = WMMA_F16(af[i], bf[j], acc[i][j]);
  }

#pragma unroll
  for (int i = 0; i < 2; ++i)
#pragma unroll
    for (int j = 0; j < 2; ++j)
#pragma unroll
      for (int r = 0; r < 8; ++r) {
        int c = wm * 32 + i * 16 + half * 8 + r;
        int n = nB + wn * 32 + j * 16 + l16;
        if (c < 48)
          outh[((size_t)b * 384 + h * 48 + c) * 4096 + n] = (_Float16)acc[i][j][r];
      }
}

// ---------------------------------------------------------------------------
extern "C" void kernel_launch(void* const* d_in, const int* in_sizes, int n_in,
                              void* d_out, int out_size, void* d_ws, size_t ws_size,
                              hipStream_t stream) {
  const float* x     = (const float*)d_in[0];  // (16,384,64,64)
  const float* W_qkv = (const float*)d_in[1];  // (1152,384)
  const float* b_qkv = (const float*)d_in[2];  // (1152,)
  const float* W_dw  = (const float*)d_in[3];  // (1152,1,3,3)
  const float* b_dw  = (const float*)d_in[4];  // (1152,)
  const float* W_out = (const float*)d_in[5];  // (384,384)
  const float* b_out = (const float*)d_in[6];  // (384,)
  const float* temp  = (const float*)d_in[7];  // (8,1,1)
  float* out = (float*)d_out;

  char* ws = (char*)d_ws;
  // Arena (lifetime-overlapped):
  _Float16* xt    = (_Float16*)(ws + 0);           // 50,331,648 B; reused as outh
  _Float16* wqh   = (_Float16*)(ws + 50331648);    //    884,736 B
  _Float16* woh   = (_Float16*)(ws + 51216384);    //    294,912 B
  _Float16* qkvh  = (_Float16*)(ws + 51511296);    // 150,994,944 B; reused for attn
  _Float16* dwh   = (_Float16*)(ws + 202506240);   // 150,994,944 B
  float*    attnf = (float*)(ws + 51511296);       //   1,179,648 B (reuses qkvh)
  _Float16* attnh = (_Float16*)(ws + 52690944);    //   1,048,576 B
  _Float16* outh  = xt;                            // xt dead after QKV GEMM

  // 1) weight conversions + x transpose-to-[n][k] f16
  k_cvt_f16<<<dim3((442368 + 255) / 256), dim3(256), 0, stream>>>(W_qkv, wqh, 442368);
  k_cvt_f16<<<dim3((147456 + 255) / 256), dim3(256), 0, stream>>>(W_out, woh, 147456);
  k_transpose<<<dim3(64, 6, 16), dim3(256), 0, stream>>>(x, xt);

  // 2) QKV GEMM (async double-buffered), +b_qkv, f16 out [b][1152][4096]
  k_gemm_qkv<<<dim3(32, 9, 16), dim3(256), 40960, stream>>>(wqh, xt, b_qkv, qkvh);
  // 3) depthwise 3x3 + b_dw
  k_dwconv<<<dim3(294912), dim3(256), 0, stream>>>(qkvh, W_dw, b_dw, dwh);
  // 4) L2-normalize q,k rows in place
  k_l2norm<<<dim3(12288), dim3(256), 0, stream>>>(dwh);
  // 5) attn = temp * q.k^T  (per b,head)
  k_attn_qk<<<dim3(128), dim3(128), 0, stream>>>(dwh, temp, attnf);
  // 6) softmax -> zero-padded f16 64x64 tiles
  k_softmax48<<<dim3(16 * 8 * 64), dim3(64), 0, stream>>>(attnf, attnh);
  // 7) out = attn.v
  k_attn_v<<<dim3(32, 128), dim3(256), 0, stream>>>(attnh, dwh, outh);
  // 8) output 1x1 conv GEMM, f32 out, +b_out
  k_gemm_ck<true><<<dim3(32, 3, 16), dim3(256), 0, stream>>>(woh, outh, b_out, out,
                                                             384, 4096, 384);
}
